// head_39316130627632
// MI455X (gfx1250) — matrix operand order repaired and checked by reading, not compiled, not run on hardware
//
#include <hip/hip_runtime.h>

typedef _Float16 v16h __attribute__((ext_vector_type(16)));
typedef _Float16 v8h  __attribute__((ext_vector_type(8)));
typedef float    v8f  __attribute__((ext_vector_type(8)));
typedef float    v4f  __attribute__((ext_vector_type(4)));

#define B_  16
#define T_  2048
#define C_  1024
#define H_  64

static __device__ __forceinline__ v16h cat8(v8h lo, v8h hi) {
  return __builtin_shufflevector(lo, hi, 0,1,2,3,4,5,6,7,8,9,10,11,12,13,14,15);
}

// reductions within 16-lane halves (xor of bits 0..3 never crosses lane 16)
static __device__ __forceinline__ float red_max16(float v) {
  #pragma unroll
  for (int m = 1; m <= 8; m <<= 1) v = fmaxf(v, __shfl_xor(v, m, 32));
  return v;
}
static __device__ __forceinline__ float red_sum16(float v) {
  #pragma unroll
  for (int m = 1; m <= 8; m <<= 1) v += __shfl_xor(v, m, 32);
  return v;
}

// --- CDNA5 async global->LDS (ASYNCcnt-tracked, no VGPR data round-trip) ----
// AS3->generic addrspacecast places the LDS byte offset in the low 32 bits
// (flat-LDS mapping uses addr[31:0]), so truncation recovers the DS address.
static __device__ __forceinline__ uint32_t lds_off(const void* p) {
  return (uint32_t)(uintptr_t)p;
}
static __device__ __forceinline__ void async_b128(uint32_t ldsaddr,
                                                  const _Float16* g) {
  asm volatile("global_load_async_to_lds_b128 %0, %1, off"
               :: "v"(ldsaddr), "v"(g) : "memory");
}
static __device__ __forceinline__ void wait_async0() {
  asm volatile("s_wait_asynccnt 0x0" ::: "memory");
}

// ---------------------------------------------------------------------------
// Kernel 1: q = xWq, k = xWk, v = xWv.  q,k stored f16 [B*T][64];
// v stored TRANSPOSED f16 [B][64][T] so attention V-fragments are contiguous.
// ---------------------------------------------------------------------------
__global__ __launch_bounds__(256) void proj_qkv(
    const float* __restrict__ x, const float* __restrict__ Wq,
    const float* __restrict__ Wk, const float* __restrict__ Wv,
    _Float16* __restrict__ qf, _Float16* __restrict__ kf,
    _Float16* __restrict__ vt)
{
  __shared__ _Float16 wt[3 * 64 * 128];   // [mat][n][c_local] f16, transposed W chunk
  const int tid  = threadIdx.x;
  const int wave = tid >> 5;
  const int lane = tid & 31;
  const int hl   = lane >> 4;
  const int ln   = lane & 15;
  const int row0 = blockIdx.x * 128 + wave * 16;
  const float* Wm[3] = {Wq, Wk, Wv};

  v8f acc[3][4];
  #pragma unroll
  for (int m = 0; m < 3; ++m)
    #pragma unroll
    for (int nt = 0; nt < 4; ++nt)
      #pragma unroll
      for (int j = 0; j < 8; ++j) acc[m][nt][j] = 0.0f;

  const float* xr = x + (size_t)(row0 + ln) * C_;

  for (int kc = 0; kc < C_ / 128; ++kc) {
    __syncthreads();
    // stage W chunk (128 c x 64 n), transposed + converted to f16
    for (int m = 0; m < 3; ++m) {
      const float* wsrc = Wm[m] + kc * 128 * 64;
      for (int idx = tid; idx < 128 * 16; idx += 256) {
        int c  = idx >> 4;          // 0..127
        int nq = idx & 15;          // n quad
        v4f w4 = *(const v4f*)(wsrc + c * 64 + nq * 4);
        #pragma unroll
        for (int j = 0; j < 4; ++j)
          wt[m * 8192 + (nq * 4 + j) * 128 + c] = (_Float16)w4[j];
      }
    }
    __syncthreads();

    #pragma unroll
    for (int step = 0; step < 4; ++step) {
      // A fragment: x rows, f32 -> f16.  lanes<16: K {0..7,16..23}, else {8..15,24..31}
      const int c0 = kc * 128 + step * 32;
      const float* pa = xr + c0 + (hl ? 8 : 0);
      v4f a0 = *(const v4f*)(pa);
      v4f a1 = *(const v4f*)(pa + 4);
      v4f a2 = *(const v4f*)(pa + 16);
      v4f a3 = *(const v4f*)(pa + 20);
      v16h afrag;
      #pragma unroll
      for (int j = 0; j < 4; ++j) {
        afrag[j]      = (_Float16)a0[j];
        afrag[4 + j]  = (_Float16)a1[j];
        afrag[8 + j]  = (_Float16)a2[j];
        afrag[12 + j] = (_Float16)a3[j];
      }
      const int cl = step * 32 + (hl ? 8 : 0);  // B: K {0..7,16..23} / {8..15,24..31} per half
      #pragma unroll
      for (int m = 0; m < 3; ++m) {
        // group the 4 B-fragment loads so one dscnt wait covers a WMMA burst
        v16h bf[4];
        #pragma unroll
        for (int nt = 0; nt < 4; ++nt) {
          const _Float16* wrow = wt + m * 8192 + (nt * 16 + ln) * 128 + cl;
          bf[nt] = cat8(*(const v8h*)wrow, *(const v8h*)(wrow + 16));
        }
        #pragma unroll
        for (int nt = 0; nt < 4; ++nt)
          acc[m][nt] = __builtin_amdgcn_wmma_f32_16x16x32_f16(
              false, afrag, false, bf[nt], (short)0, acc[m][nt], false, false);
      }
    }
  }

  const int bb = row0 >> 11;     // row0 / T (workgroup never straddles a batch)
  #pragma unroll
  for (int nt = 0; nt < 4; ++nt) {
    #pragma unroll
    for (int v = 0; v < 8; ++v) {
      int r = row0 + v + 8 * hl;               // C/D: lanes<16 M=v, lanes>=16 M=v+8
      int n = nt * 16 + ln;
      qf[(size_t)r * H_ + n] = (_Float16)acc[0][nt][v];
      kf[(size_t)r * H_ + n] = (_Float16)acc[1][nt][v];
      vt[((size_t)bb * H_ + n) * T_ + (r & (T_ - 1))] = (_Float16)acc[2][nt][v];
    }
  }
}

// ---------------------------------------------------------------------------
// Kernel 2: fused flash attention. 8 waves x 16 query rows. Double-buffered
// K/V tiles streamed via GLOBAL_LOAD_ASYNC_TO_LDS_B128 (overlaps HBM with
// WMMA); online softmax in f32; P bounced via per-wave LDS region.
// ---------------------------------------------------------------------------
__global__ __launch_bounds__(256) void attn_fused(
    const _Float16* __restrict__ qf, const _Float16* __restrict__ kf,
    const _Float16* __restrict__ vt, float* __restrict__ out)
{
  __shared__ _Float16 kt[2][32 * 64];   // K tiles [s][c], double buffered
  __shared__ _Float16 vl[2][64 * 32];   // V tiles [h][s] (pre-transposed)
  __shared__ _Float16 pl[8 * 16 * 32];  // per-wave P tiles [m][k]
  const int tid  = threadIdx.x;
  const int wave = tid >> 5;
  const int lane = tid & 31;
  const int hl   = lane >> 4;
  const int ln   = lane & 15;
  const int bb   = blockIdx.x >> 4;     // 16 query blocks per batch
  const int qblk = blockIdx.x & 15;
  const int q0wg = qblk * 128;
  const int q0w  = q0wg + wave * 16;

  // Q fragments: 2 contraction steps over head dim 64
  const _Float16* qrow = qf + (size_t)(bb * T_ + q0w + ln) * H_;
  v16h qfrag[2];
  #pragma unroll
  for (int f = 0; f < 2; ++f) {
    const _Float16* p = qrow + f * 32 + (hl ? 8 : 0);
    qfrag[f] = cat8(*(const v8h*)p, *(const v8h*)(p + 16));
  }

  v8f accO[4];
  #pragma unroll
  for (int nt = 0; nt < 4; ++nt)
    #pragma unroll
    for (int j = 0; j < 8; ++j) accO[nt][j] = 0.0f;
  float mst[8], lst[8];
  #pragma unroll
  for (int v = 0; v < 8; ++v) { mst[v] = -1e30f; lst[v] = 0.0f; }

  const int qhi_w  = q0w + 15;
  const int kv_end = q0wg + 127;          // uniform bound for the whole WG
  const int nIter  = kv_end / 32 + 1;
  _Float16* pbase = pl + wave * 512;

  // per-thread staging addresses (16B K chunk + 16B V chunk per thread)
  const int ks = tid >> 3, kc_ = tid & 7;     // K: row s, f16-octet kc_
  const int vh = tid >> 2, vc = tid & 3;      // V: row h, f16-octet vc
  const _Float16* kgsrc = kf + (size_t)(bb * T_ + ks) * H_ + kc_ * 8;
  const _Float16* vgsrc = vt + (size_t)(bb * H_ + vh) * T_ + vc * 8;
  uint32_t kdst[2], vdst[2];
  #pragma unroll
  for (int b = 0; b < 2; ++b) {
    kdst[b] = lds_off(&kt[b][ks * 64 + kc_ * 8]);
    vdst[b] = lds_off(&vl[b][vh * 32 + vc * 8]);
  }

  auto stage = [&](int buf, int kv0) {
    async_b128(kdst[buf], kgsrc + (size_t)kv0 * H_);
    async_b128(vdst[buf], vgsrc + kv0);
  };

  stage(0, 0);
  wait_async0();
  __syncthreads();

  for (int it = 0; it < nIter; ++it) {
    const int kv0 = it * 32;
    const int cur = it & 1;
    if (it + 1 < nIter) stage(cur ^ 1, kv0 + 32);  // overlap next tile with compute

    if (kv0 <= qhi_w) {                            // wave-uniform: EXEC all-ones
      const _Float16* ktc = kt[cur];
      const _Float16* vlc = vl[cur];

      // S = Q K^T : two 16x16 score tiles (sc = key half)
      v8f S[2];
      #pragma unroll
      for (int sc = 0; sc < 2; ++sc) {
        #pragma unroll
        for (int j = 0; j < 8; ++j) S[sc][j] = 0.0f;
        const int s = sc * 16 + ln;
        v16h bk[2];
        #pragma unroll
        for (int f = 0; f < 2; ++f) {
          const _Float16* kr = ktc + s * 64 + f * 32 + (hl ? 8 : 0);
          bk[f] = cat8(*(const v8h*)kr, *(const v8h*)(kr + 16));
        }
        #pragma unroll
        for (int f = 0; f < 2; ++f)
          S[sc] = __builtin_amdgcn_wmma_f32_16x16x32_f16(
              false, qfrag[f], false, bk[f], (short)0, S[sc], false, false);
      }
      // scale by C^-0.5 (reference scales by n_embd, not head_size) + causal mask
      #pragma unroll
      for (int sc = 0; sc < 2; ++sc) {
        const int key = kv0 + sc * 16 + ln;
        #pragma unroll
        for (int v = 0; v < 8; ++v) {
          float sv = S[sc][v] * 0.03125f;
          int q = q0w + v + 8 * hl;
          S[sc][v] = (key > q) ? -1e30f : sv;
        }
      }
      // online softmax (joint over both halves so P/l/acc share one max)
      float p0[8], p1[8];
      #pragma unroll
      for (int v = 0; v < 8; ++v) {
        float mt   = red_max16(fmaxf(S[0][v], S[1][v]));
        float mn   = fmaxf(mst[v], mt);
        float corr = __expf(mst[v] - mn);
        p0[v] = __expf(S[0][v] - mn);
        p1[v] = __expf(S[1][v] - mn);
        float rs = red_sum16(p0[v] + p1[v]);
        lst[v] = lst[v] * corr + rs;
        mst[v] = mn;
        #pragma unroll
        for (int nt = 0; nt < 4; ++nt) accO[nt][v] *= corr;
      }
      // bounce P through per-wave LDS: C-layout -> A-layout (row-major [16][32])
      #pragma unroll
      for (int v = 0; v < 8; ++v) {
        int m = v + 8 * hl;
        pbase[m * 32 + ln]      = (_Float16)p0[v];
        pbase[m * 32 + 16 + ln] = (_Float16)p1[v];
      }
      const _Float16* pr = pbase + ln * 32 + (hl ? 8 : 0);
      v16h pfrag = cat8(*(const v8h*)pr, *(const v8h*)(pr + 16));
      // O += P V : group the 4 V fragments, then 4 back-to-back WMMAs
      v16h bv[4];
      #pragma unroll
      for (int nt = 0; nt < 4; ++nt) {
        const _Float16* vr = vlc + (nt * 16 + ln) * 32 + (hl ? 8 : 0);
        bv[nt] = cat8(*(const v8h*)vr, *(const v8h*)(vr + 16));
      }
      #pragma unroll
      for (int nt = 0; nt < 4; ++nt)
        accO[nt] = __builtin_amdgcn_wmma_f32_16x16x32_f16(
            false, pfrag, false, bv[nt], (short)0, accO[nt], false, false);
    }

    wait_async0();       // next tile fully in LDS
    __syncthreads();     // all waves done reading cur before it is re-staged
  }

  // epilogue: out = accO / l
  #pragma unroll
  for (int nt = 0; nt < 4; ++nt) {
    #pragma unroll
    for (int v = 0; v < 8; ++v) {
      int r = q0w + v + 8 * hl;
      out[(size_t)(bb * T_ + r) * H_ + nt * 16 + ln] = accO[nt][v] / lst[v];
    }
  }
}

extern "C" void kernel_launch(void* const* d_in, const int* in_sizes, int n_in,
                              void* d_out, int out_size, void* d_ws, size_t ws_size,
                              hipStream_t stream) {
  (void)in_sizes; (void)n_in; (void)out_size; (void)ws_size;
  const float* x  = (const float*)d_in[0];
  const float* Wq = (const float*)d_in[1];
  const float* Wk = (const float*)d_in[2];
  const float* Wv = (const float*)d_in[3];
  // workspace: q f16 (4MB) | k f16 (4MB) | v-transposed f16 (4MB)
  _Float16* qf = (_Float16*)d_ws;
  _Float16* kf = qf + (size_t)B_ * T_ * H_;
  _Float16* vt = kf + (size_t)B_ * T_ * H_;
  proj_qkv<<<(B_ * T_) / 128, 256, 0, stream>>>(x, Wq, Wk, Wv, qf, kf, vt);
  attn_fused<<<(B_ * T_) / 128, 256, 0, stream>>>(qf, kf, vt, (float*)d_out);
}
